// CrossModalAttention_16466904613743
// MI455X (gfx1250) — matrix operand order repaired and checked by reading, not compiled, not run on hardware
//
#include <hip/hip_runtime.h>
#include <hip/hip_bf16.h>

typedef __attribute__((ext_vector_type(16))) _Float16 v16h;
typedef __attribute__((ext_vector_type(8)))  float    v8f;
typedef __attribute__((ext_vector_type(4)))  float    f32x4;
typedef __attribute__((ext_vector_type(4)))  unsigned u32x4;

// TDM descriptor vector types (probe-verified arities in CDNA5_HIP.md)
typedef __attribute__((ext_vector_type(4))) unsigned tdm_v4u;
typedef __attribute__((ext_vector_type(8))) int      tdm_v8i;
typedef __attribute__((ext_vector_type(4))) int      tdm_v4i;

#define C_DIM   256
#define W_DIM   64
#define HW      4096                 // 64*64
#define IMG     (C_DIM * HW)         // floats per image = 1048576
#define KSTEP   32
#define LDS_PITCH 20                 // dwords per 32-half px row (16 used + 4 pad) = 80B

// --------- staging-path selection (compile-time, always falls back clean) ----
#if __has_builtin(__builtin_amdgcn_tensor_load_to_lds)
#define USE_TDM 1
#elif __has_builtin(__builtin_amdgcn_global_load_async_to_lds_b128)
#define USE_ASYNC 1
#endif

// =====================================================================
// Prep kernel A: convert blue/white fp32 [b][c][h][w]  ->  f16 [b][p][c]
// (pixel-major rows so GEMM A-tiles are contiguous 64B rows per pixel).
// grid (64 px-blocks, 8 b, 2 src), 256 threads. LDS tile transpose.
// =====================================================================
#define PITCH_P 132
__global__ __launch_bounds__(256)
void prep_feat_f16(const float* __restrict__ blue, const float* __restrict__ white,
                   _Float16* __restrict__ feat16)
{
    __shared__ __align__(16) unsigned ldsT[64 * PITCH_P];
    const int pbase = blockIdx.x * 64;
    const int b     = blockIdx.y;
    const int sidx  = blockIdx.z;
    const float* src = (sidx == 0 ? blue : white) + (size_t)b * IMG;
    _Float16*   dstb = feat16 + ((size_t)sidx * 8 + b) * (size_t)HW * C_DIM;

    const int t   = threadIdx.x;
    const int cp  = t & 127;          // channel pair 2cp, 2cp+1
    const int px0 = (t >> 7) * 32;    // 0 or 32

    const float* r0 = src + (size_t)(2 * cp) * HW + pbase + px0;
    const float* r1 = r0 + HW;
    union { f32x4 v[8]; float f[32]; } ua, ub;
#pragma unroll
    for (int q = 0; q < 8; ++q) { ua.v[q] = *(const f32x4*)(r0 + 4 * q); ub.v[q] = *(const f32x4*)(r1 + 4 * q); }
#pragma unroll
    for (int j = 0; j < 32; ++j) {
        union { _Float16 h[2]; unsigned u; } pk;
        pk.h[0] = (_Float16)ua.f[j];
        pk.h[1] = (_Float16)ub.f[j];
        ldsT[(px0 + j) * PITCH_P + cp] = pk.u;
    }
    __syncthreads();
    const int px  = t >> 2;           // 0..63
    const int seg = t & 3;            // 128B segment of the 512B row
    const unsigned* lr = &ldsT[px * PITCH_P + seg * 32];
    unsigned* gw = (unsigned*)(dstb + (size_t)(pbase + px) * C_DIM) + seg * 32;
#pragma unroll
    for (int q = 0; q < 8; ++q)
        *(u32x4*)(gw + 4 * q) = *(const u32x4*)(lr + 4 * q);
}

// =====================================================================
// Prep kernel B: weights fp32 -> f16, same [o][c] row-major layout.
// 3*256*256 elements, 8 per thread -> 96 blocks of 256.
// =====================================================================
__global__ __launch_bounds__(256)
void prep_weights_f16(const float* __restrict__ Wq, const float* __restrict__ Wk,
                      const float* __restrict__ Wv, _Float16* __restrict__ wh)
{
    const int g    = blockIdx.x * 256 + threadIdx.x;  // 0..24575
    const int proj = g >> 13;
    const int idx8 = (g & 8191) * 8;
    const float* Wm = (proj == 0) ? Wq : (proj == 1 ? Wk : Wv);
    union { f32x4 v[2]; float f[8]; } u;
    u.v[0] = *(const f32x4*)(Wm + idx8);
    u.v[1] = *(const f32x4*)(Wm + idx8 + 4);
    union { _Float16 h[8]; u32x4 q; } o;
#pragma unroll
    for (int e = 0; e < 8; ++e) o.h[e] = (_Float16)u.f[e];
    *(u32x4*)(wh + (size_t)proj * 65536 + idx8) = o.q;
}

// =====================================================================
// Kernel 1: Q/K/V projection GEMM, V_WMMA_F32_16X16X32_F16, f32 accum.
// A-tile (256px x 32ch f16, 80B-pitch rows) staged into LDS by the
// Tensor Data Mover (preferred), async-to-LDS, or a sync copy fallback.
// No converts in the hot loop: everything is pre-cast f16.
// =====================================================================
__global__ __launch_bounds__(256)
void qkv_proj_wmma(const _Float16* __restrict__ feat16,
                   const _Float16* __restrict__ wh16,
                   const float* __restrict__ bq, const float* __restrict__ bk,
                   const float* __restrict__ bv,
                   float* __restrict__ qkv)
{
    __shared__ __align__(16) unsigned int ldsA[256 * LDS_PITCH];

    const int proj  = blockIdx.z;
    const int b     = blockIdx.y >> 2;
    const int nblk  = blockIdx.y & 3;
    const int pbase = blockIdx.x * 256;
    const int nbase = nblk * 64;
    const int sidx  = (proj == 0) ? 0 : 1;   // blue for Q, white for K/V

    const _Float16* srcb16 = feat16 + ((size_t)sidx * 8 + b) * (size_t)HW * C_DIM;
    const _Float16* Wm16   = wh16 + (size_t)proj * 65536;
    const float*    bias   = (proj == 0) ? bq : (proj == 1 ? bk : bv);
    float*          dst    = qkv + (size_t)proj * 8 * IMG + (size_t)b * IMG;

    const int tid   = threadIdx.x;
    const int lane  = tid & 31;
    const int wv_id = tid >> 5;
    const int m     = lane & 15;
    const int s     = lane >> 4;

    v8f acc[2][4];
#pragma unroll
    for (int i = 0; i < 2; ++i)
#pragma unroll
        for (int j = 0; j < 4; ++j)
#pragma unroll
            for (int r = 0; r < 8; ++r) acc[i][j][r] = 0.0f;

    for (int kb = 0; kb < C_DIM; kb += KSTEP) {
        // ---- stage A tile: feat16[pbase..+255][kb..kb+31] -> LDS [px][c] ----
#if defined(USE_TDM)
        if (wv_id == 0) {
            // Tensor DMA descriptor (CDNA5 ISA): 2-D tile,
            // 2B elements, tile 32 x 256, row stride 256, LDS pad 4dw / 16dw.
            const unsigned lds_base = (unsigned)(uintptr_t)&ldsA[0];
            const unsigned long long ga =
                (unsigned long long)(uintptr_t)(srcb16 + ((size_t)pbase * C_DIM + kb));
            tdm_v4u g0;
            g0[0] = 1u;                                        // count=1 (valid D#)
            g0[1] = lds_base;                                  // lds_addr
            g0[2] = (unsigned)(ga & 0xffffffffu);              // global_addr lo
            g0[3] = (unsigned)((ga >> 32) & 0x01ffffffu) | (2u << 30); // hi | type=2
            tdm_v8i g1;
            g1[0] = (int)((1u << 16)   // data_size = 2B
                        | (1u << 20)   // pad_enable
                        | (3u << 22)   // pad_interval: 16 dwords
                        | (3u << 25)); // pad_amount:   4 dwords
            g1[1] = (int)(32u << 16);  // tensor_dim0 = 32     (bits 79:48 lo)
            g1[2] = (int)(256u << 16); // tensor_dim1 = 256    (bits 111:80 lo)
            g1[3] = (int)(32u << 16);  // tile_dim0  = 32      (bits 127:112)
            g1[4] = 256;               // tile_dim1  = 256, tile_dim2 = 0
            g1[5] = 256;               // tensor_dim0_stride = 256 elements
            g1[6] = 0;
            g1[7] = 0;
            tdm_v4i g2; g2[0] = 1; g2[1] = 1; g2[2] = 0; g2[3] = 0;            // dim2=1, dim3=1
            tdm_v4i g3; g3[0] = 0; g3[1] = (int)(1u << 16); g3[2] = 0; g3[3] = 0; // dim4=1
#if __clang_major__ >= 23
            tdm_v8i gz = 0;
            __builtin_amdgcn_tensor_load_to_lds(g0, g1, g2, g3, gz, 0);
#else
            __builtin_amdgcn_tensor_load_to_lds(g0, g1, g2, g3, 0);
#endif
            __builtin_amdgcn_s_wait_tensorcnt(0);
        }
        __syncthreads();
#elif defined(USE_ASYNC)
        {
            typedef __attribute__((address_space(1))) const char gchar;
            typedef __attribute__((address_space(3))) char lchar;
            const _Float16* gsrc = srcb16 + ((size_t)(pbase + tid) * C_DIM + kb);
            gchar* gp = (gchar*)(uintptr_t)gsrc;
            lchar* lp = (lchar*)(unsigned)(uintptr_t)&ldsA[tid * LDS_PITCH];
#pragma unroll
            for (int g4 = 0; g4 < 4; ++g4)
                __builtin_amdgcn_global_load_async_to_lds_b128(gp + 16 * g4, lp + 16 * g4, 0, 0);
        }
#if __has_builtin(__builtin_amdgcn_s_wait_asynccnt)
        __builtin_amdgcn_s_wait_asynccnt(0);
#else
        asm volatile("s_wait_asynccnt 0" ::: "memory");
#endif
        __syncthreads();
#else
        {
            const u32x4* gsrc = (const u32x4*)(srcb16 + ((size_t)(pbase + tid) * C_DIM + kb));
            if (kb + KSTEP < C_DIM)
                __builtin_prefetch(srcb16 + ((size_t)(pbase + tid) * C_DIM + kb + KSTEP), 0, 0);
            unsigned* lrow = &ldsA[tid * LDS_PITCH];
            u32x4 t0 = gsrc[0], t1 = gsrc[1], t2 = gsrc[2], t3 = gsrc[3];
            *(u32x4*)(lrow)      = t0;
            *(u32x4*)(lrow + 4)  = t1;
            *(u32x4*)(lrow + 8)  = t2;
            *(u32x4*)(lrow + 12) = t3;
        }
        __syncthreads();
#endif

        // ---- B fragments: pre-cast f16 weights, B[k][n] = W[o=n][kb+k] ----
        v16h bfrag[4];
#pragma unroll
        for (int j = 0; j < 4; ++j) {
            const int o = nbase + j * 16 + m;
            const _Float16* wr = Wm16 + (size_t)o * C_DIM + kb + s * 8;
            union { u32x4 u[2]; v16h h; } fb;
            fb.u[0] = *(const u32x4*)(wr);
            fb.u[1] = *(const u32x4*)(wr + 16);
            bfrag[j] = fb.h;
        }

        // ---- A fragments (two ds_load_b128 each) + WMMA ----
#pragma unroll
        for (int i = 0; i < 2; ++i) {
            const int px = (2 * wv_id + i) * 16 + m;
            const unsigned dwbase = (unsigned)px * LDS_PITCH + s * 4;
            union { u32x4 u[2]; v16h h; } fa;
            fa.u[0] = *reinterpret_cast<const u32x4*>(&ldsA[dwbase]);     // c = s*8..
            fa.u[1] = *reinterpret_cast<const u32x4*>(&ldsA[dwbase + 8]); // c = 16+s*8..
#pragma unroll
            for (int j = 0; j < 4; ++j)
                acc[i][j] = __builtin_amdgcn_wmma_f32_16x16x32_f16(
                    false, fa.h, false, bfrag[j], (short)0, acc[i][j], false, false);
        }
        __syncthreads();
    }

    // ---- epilogue: bias + store Y in [b][c][h][w] fp32 layout ----
#pragma unroll
    for (int i = 0; i < 2; ++i) {
        const int px = pbase + (2 * wv_id + i) * 16 + s * 8;
#pragma unroll
        for (int j = 0; j < 4; ++j) {
            const int o = nbase + j * 16 + m;
            const float bo = bias[o];
            float* dp = dst + (size_t)o * HW + px;
            f32x4 lo, hi;
#pragma unroll
            for (int r = 0; r < 4; ++r) { lo[r] = acc[i][j][r] + bo; hi[r] = acc[i][j][4 + r] + bo; }
            *(f32x4*)(dp)     = lo;
            *(f32x4*)(dp + 4) = hi;
        }
    }
}

// =====================================================================
// Kernel 2: per-channel 3x3-token neighborhood attention + residual.
// WG per token; thread = channel; zero-padded OOB neighbors stay in the
// softmax denominator (logit 0, V 0), matching the reference exactly.
// =====================================================================
__global__ __launch_bounds__(256)
void neighbor_token_attn(const float* __restrict__ blue,
                         const float* __restrict__ qkv,
                         float* __restrict__ out)
{
    const int c  = threadIdx.x;
    const int tk = blockIdx.x;
    const int b  = tk >> 8;
    const int ti = (tk >> 4) & 15;
    const int tj = tk & 15;

    const size_t chan = (size_t)b * IMG + (size_t)c * HW;
    const float* Qp = qkv + chan;
    const float* Kp = qkv + (size_t)8  * IMG + chan;
    const float* Vp = qkv + (size_t)16 * IMG + chan;
    const size_t tok = (size_t)(ti * 4) * W_DIM + tj * 4;

    float q[16];
#pragma unroll
    for (int r = 0; r < 4; ++r) {
        f32x4 t4 = *(const f32x4*)(Qp + tok + r * W_DIM);
#pragma unroll
        for (int e = 0; e < 4; ++e) q[4 * r + e] = t4[e];
    }

    float logits[9];
    unsigned inb = 0;
#pragma unroll
    for (int n = 0; n < 9; ++n) {
        const int tii = ti + (n / 3) - 1;
        const int tjj = tj + (n % 3) - 1;
        float lg = 0.0f;
        if ((unsigned)tii < 16u && (unsigned)tjj < 16u) {
            inb |= 1u << n;
            const float* kp = Kp + (size_t)(tii * 4) * W_DIM + tjj * 4;
            float d = 0.0f;
#pragma unroll
            for (int r = 0; r < 4; ++r) {
                f32x4 k4 = *(const f32x4*)(kp + r * W_DIM);
#pragma unroll
                for (int e = 0; e < 4; ++e) d += k4[e] * q[4 * r + e];
            }
            lg = d * 0.25f;
        }
        logits[n] = lg;
    }

    float mx = logits[0];
#pragma unroll
    for (int n = 1; n < 9; ++n) mx = fmaxf(mx, logits[n]);
    float wn[9], wsum = 0.0f;
#pragma unroll
    for (int n = 0; n < 9; ++n) { wn[n] = __expf(logits[n] - mx); wsum += wn[n]; }
    const float inv = 1.0f / wsum;

    float accv[16];
#pragma unroll
    for (int e = 0; e < 16; ++e) accv[e] = 0.0f;
#pragma unroll
    for (int n = 0; n < 9; ++n) {
        if (inb & (1u << n)) {
            const int tii = ti + (n / 3) - 1;
            const int tjj = tj + (n % 3) - 1;
            const float* vp = Vp + (size_t)(tii * 4) * W_DIM + tjj * 4;
            const float wgt = wn[n];
#pragma unroll
            for (int r = 0; r < 4; ++r) {
                f32x4 v4 = *(const f32x4*)(vp + r * W_DIM);
#pragma unroll
                for (int e = 0; e < 4; ++e) accv[4 * r + e] += wgt * v4[e];
            }
        }
    }

    const float* bl = blue + chan + tok;
    float*       op = out  + chan + tok;
#pragma unroll
    for (int r = 0; r < 4; ++r) {
        f32x4 b4 = *(const f32x4*)(bl + r * W_DIM);
        f32x4 o4;
#pragma unroll
        for (int e = 0; e < 4; ++e) o4[e] = b4[e] + accv[4 * r + e] * inv;
        *(f32x4*)(op + r * W_DIM) = o4;
    }
}

extern "C" void kernel_launch(void* const* d_in, const int* in_sizes, int n_in,
                              void* d_out, int out_size, void* d_ws, size_t ws_size,
                              hipStream_t stream)
{
    (void)in_sizes; (void)n_in; (void)out_size; (void)ws_size;
    const float* blue  = (const float*)d_in[0];
    const float* white = (const float*)d_in[1];
    const float* Wq = (const float*)d_in[2];
    const float* bq = (const float*)d_in[3];
    const float* Wk = (const float*)d_in[4];
    const float* bk = (const float*)d_in[5];
    const float* Wv = (const float*)d_in[6];
    const float* bv = (const float*)d_in[7];
    float* out = (float*)d_out;

    // workspace map (all L2-resident on MI455X's 192MB L2):
    //   [0, 24*IMG) floats          : Q,K,V fp32                (100.7 MB)
    //   feat16: 2*8*HW*C halfs      : blue/white f16, [b][p][c] ( 33.6 MB)
    //   wh16:   3*C*C halfs         : Wq/Wk/Wv f16              (  0.4 MB)
    float*     qkv    = (float*)d_ws;
    _Float16*  feat16 = (_Float16*)(qkv + (size_t)24 * IMG);
    _Float16*  wh16   = feat16 + (size_t)2 * 8 * HW * C_DIM;

    prep_feat_f16   <<<dim3(64, 8, 2), 256, 0, stream>>>(blue, white, feat16);
    prep_weights_f16<<<dim3(96),       256, 0, stream>>>(Wq, Wk, Wv, wh16);
    qkv_proj_wmma   <<<dim3(16, 32, 3), 256, 0, stream>>>(feat16, wh16, bq, bk, bv, qkv);
    neighbor_token_attn<<<dim3(2048),  256, 0, stream>>>(blue, qkv, out);
}
